// MultiHeadAttention_29901562314894
// MI455X (gfx1250) — matrix operand order repaired and checked by reading, not compiled, not run on hardware
//
#include <hip/hip_runtime.h>
#include <hip/hip_bf16.h>

// MI455X / gfx1250, wave32. Matrix math on v_wmma_f32_16x16x32_bf16.
// Tile staging: TDM (tensor_load_to_lds, TENSORcnt) when the builtin exists,
// else CDNA5 async-to-LDS (ASYNCcnt). Both are gfx1250-specific paths.

typedef __attribute__((ext_vector_type(16))) __bf16 v16bf;
typedef __attribute__((ext_vector_type(8)))  float  v8f;

#define D_MODEL 1024
#define NHEAD   16
#define SEQ     2048
#define HDIM    64

#define BM 128
#define BN 128
#define BK 32

// ---- CDNA5 async global->LDS copy (16B per lane), tracked by ASYNCcnt ------
__device__ __forceinline__ void async_ld_b128(const void* g, void* l) {
  asm volatile("global_load_async_to_lds_b128 %0, %1, off"
               :
               : "v"((unsigned int)(uintptr_t)l),
                 "v"((unsigned long long)(uintptr_t)g)
               : "memory");
}
__device__ __forceinline__ void wait_async0() {
  asm volatile("s_wait_asynccnt 0x0" ::: "memory");
}
__device__ __forceinline__ void wait_tensor0() {
  asm volatile("s_wait_tensorcnt 0x0" ::: "memory");
}

// ---- Tensor Data Mover: one instruction stages a whole strided 2D tile -----
#if __has_builtin(__builtin_amdgcn_tensor_load_to_lds)
#define HAVE_TDM 1
typedef __attribute__((ext_vector_type(4))) unsigned int u32x4;
typedef __attribute__((ext_vector_type(8))) int i32x8;
typedef __attribute__((ext_vector_type(4))) int i32x4;

// Load tile_h rows x tile_w bf16 elements (row stride = row_stride elems in
// global memory) into LDS, rows packed contiguously. D# layout per CDNA5 ISA
// ch.8 (group0: flags/lds/global addr; group1: dims/tile/stride).
__device__ __forceinline__ void tdm_load_2d(void* lds, const void* gsrc,
                                            unsigned tile_w, unsigned tile_h,
                                            unsigned row_stride) {
  const unsigned lds_addr = (unsigned)(uintptr_t)lds;
  const unsigned long long ga = (unsigned long long)(uintptr_t)gsrc;
  u32x4 g0;
  g0[0] = 1u;                                      // count=1, user descriptor
  g0[1] = lds_addr;                                // lds_addr (bytes)
  g0[2] = (unsigned)ga;                            // global_addr[31:0]
  g0[3] = (unsigned)((ga >> 32) & 0x01FFFFFFull)   // global_addr[56:32]
          | 0x80000000u;                           // type=2 ("image")
  i32x8 g1;
  g1[0] = (int)(1u << 16);                                       // data_size=2B
  g1[1] = (int)((tile_w & 0xFFFFu) << 16);                       // tensor_dim0 lo
  g1[2] = (int)((tile_w >> 16) | ((tile_h & 0xFFFFu) << 16));    // td0 hi|td1 lo
  g1[3] = (int)((tile_h >> 16) | (tile_w << 16));                // td1 hi|tile_dim0
  g1[4] = (int)(tile_h & 0xFFFFu);                               // tile_dim1
  g1[5] = (int)row_stride;                                       // dim0_stride lo
  g1[6] = 0; g1[7] = 0;
  i32x4 g2; g2[0] = 1; g2[1] = 1; g2[2] = 0; g2[3] = 0;  // dim2/3=1, tile_dim3=0
  i32x4 g3; g3[0] = 0; g3[1] = 1 << 16; g3[2] = 0; g3[3] = 0;  // dim4=1
  i32x8 g4 = {0, 0, 0, 0, 0, 0, 0, 0};  // unused trailing group (6-arg form)
  __builtin_amdgcn_tensor_load_to_lds(g0, g1, g2, g3, g4, 0);
}
#else
#define HAVE_TDM 0
#endif

__device__ __forceinline__ v8f wmma_bf16(v16bf a, v16bf b, v8f c) {
  return __builtin_amdgcn_wmma_f32_16x16x32_bf16(false, a, false, b, (short)0, c,
                                                 false, false);
}

// A-matrix fragment, 16x32 bf16, source row-major [m][k].
__device__ __forceinline__ v16bf frag_a16(const __bf16* base, int stride) {
  const int lane = threadIdx.x & 31;
  const __bf16* p = base + (lane & 15) * stride + ((lane < 16) ? 0 : 8);
  union { v16bf v; uint4 u[2]; } t;
  t.u[0] = *(const uint4*)(p);
  t.u[1] = *(const uint4*)(p + 16);
  return t.v;
}

// B-matrix fragment, 32x16 bf16, source stored [n][k] (K contiguous per col).
__device__ __forceinline__ v16bf frag_b16(const __bf16* base, int stride) {
  const int lane = threadIdx.x & 31;
  const __bf16* p = base + (lane & 15) * stride + ((lane < 16) ? 0 : 8);
  union { v16bf v; uint4 u[2]; } t;
  t.u[0] = *(const uint4*)(p);
  t.u[1] = *(const uint4*)(p + 16);
  return t.v;
}

// ---------------------------------------------------------------- convert ---
__global__ __launch_bounds__(256) void cvt_f32_bf16(const float* __restrict__ in,
                                                    void* __restrict__ outv,
                                                    int n4) {
  const int i = blockIdx.x * 256 + threadIdx.x;
  if (i < n4) {
    const float4 f = ((const float4*)in)[i];
    union { __bf16 h[4]; uint2 u; } t;
    t.h[0] = (__bf16)f.x; t.h[1] = (__bf16)f.y;
    t.h[2] = (__bf16)f.z; t.h[3] = (__bf16)f.w;
    ((uint2*)outv)[i] = t.u;
  }
}

// ------------------------------------------------------------------- GEMM ---
// out[m,n] = sum_k A[m,k] * Wt[n,k] + bias[n]
// mode 0: bf16 scattered to [B, H, S, DK];  mode 1: f32 row-major [M, N].
#if !HAVE_TDM
__device__ __forceinline__ void stage_tile_async(
    const __bf16* __restrict__ A, const __bf16* __restrict__ Bw, int Kdim,
    int m0, int n0, int k0, __bf16* as, __bf16* bs, int tid) {
#pragma unroll
  for (int i = tid; i < (BM * BK) / 8; i += 256) {
    const int r = i >> 2, c = i & 3;
    async_ld_b128(A + (size_t)(m0 + r) * Kdim + k0 + c * 8, as + r * BK + c * 8);
  }
#pragma unroll
  for (int i = tid; i < (BN * BK) / 8; i += 256) {
    const int r = i >> 2, c = i & 3;
    async_ld_b128(Bw + (size_t)(n0 + r) * Kdim + k0 + c * 8, bs + r * BK + c * 8);
  }
}
#endif

__global__ __launch_bounds__(256) void gemm_wmma(
    const void* __restrict__ Av, const void* __restrict__ Bv,
    const float* __restrict__ bias, void* __restrict__ outv,
    int Kdim, int Ndim, int mode) {
  const __bf16* A  = (const __bf16*)Av;
  const __bf16* Bw = (const __bf16*)Bv;

  __shared__ __attribute__((aligned(16))) __bf16 As[2][BM * BK];  // [m][k]
  __shared__ __attribute__((aligned(16))) __bf16 Bs[2][BN * BK];  // [n][k]

  const int tid  = threadIdx.x;
  const int lane = tid & 31;
  const int wave = tid >> 5;
  const int wm = wave & 3;   // 4 wave-rows  x 2 mtiles
  const int wn = wave >> 2;  // 2 wave-cols  x 4 ntiles
  const int m0 = blockIdx.y * BM;
  const int n0 = blockIdx.x * BN;

  v8f acc[2][4] = {};
  const int nk = Kdim / BK;

#if HAVE_TDM
  if (wave == 0) {
    tdm_load_2d(As[0], A + (size_t)m0 * Kdim, BK, BM, Kdim);
    tdm_load_2d(Bs[0], Bw + (size_t)n0 * Kdim, BK, BN, Kdim);
  }
#else
  stage_tile_async(A, Bw, Kdim, m0, n0, 0, As[0], Bs[0], tid);
#endif

  for (int t = 0; t < nk; ++t) {
    const int cur = t & 1;
#if HAVE_TDM
    if (wave == 0) wait_tensor0();  // buf[cur] tiles landed in LDS
#else
    wait_async0();
#endif
    __syncthreads();                // visible to all; buf[cur^1] readers done
    if (t + 1 < nk) {
#if HAVE_TDM
      if (wave == 0) {
        tdm_load_2d(As[cur ^ 1], A + (size_t)m0 * Kdim + (t + 1) * BK, BK, BM, Kdim);
        tdm_load_2d(Bs[cur ^ 1], Bw + (size_t)n0 * Kdim + (t + 1) * BK, BK, BN, Kdim);
      }
#else
      stage_tile_async(A, Bw, Kdim, m0, n0, (t + 1) * BK, As[cur ^ 1],
                       Bs[cur ^ 1], tid);
#endif
    }

    v16bf af[2], bf[4];
#pragma unroll
    for (int i = 0; i < 2; ++i)
      af[i] = frag_a16(As[cur] + (wm * 2 + i) * 16 * BK, BK);
#pragma unroll
    for (int j = 0; j < 4; ++j)
      bf[j] = frag_b16(Bs[cur] + (wn * 4 + j) * 16 * BK, BK);
#pragma unroll
    for (int i = 0; i < 2; ++i)
#pragma unroll
      for (int j = 0; j < 4; ++j)
        acc[i][j] = wmma_bf16(af[i], bf[j], acc[i][j]);
  }

  // Epilogue: C layout = col (lane&15), rows r + (lane<16?0:8).
  const int halfrow = (lane < 16) ? 0 : 8;
#pragma unroll
  for (int i = 0; i < 2; ++i) {
#pragma unroll
    for (int j = 0; j < 4; ++j) {
      const int col   = n0 + (wn * 4 + j) * 16 + (lane & 15);
      const float bval = bias[col];
      const int mrow0 = m0 + (wm * 2 + i) * 16 + halfrow;
#pragma unroll
      for (int r = 0; r < 8; ++r) {
        const int row = mrow0 + r;
        const float vv = acc[i][j][r] + bval;
        if (mode == 0) {
          const int b = row >> 11, s = row & (SEQ - 1);
          const int h = col >> 6,  dk = col & (HDIM - 1);
          ((__bf16*)outv)[(((size_t)(b * NHEAD + h) * SEQ + s) << 6) + dk] =
              (__bf16)vv;
        } else {
          ((float*)outv)[(size_t)row * Ndim + col] = vv;
        }
      }
    }
  }
}

// ------------------------------------------------------- flash attention ----
// Grid: (SEQ/128, B*H). 8 waves, each wave owns 16 query rows.
__global__ __launch_bounds__(256) void attn_fmha(
    const void* __restrict__ Qv, const void* __restrict__ Kv,
    const void* __restrict__ Vv, const int* __restrict__ mask,
    void* __restrict__ Ov) {
  const __bf16* Q = (const __bf16*)Qv;
  const __bf16* K = (const __bf16*)Kv;
  const __bf16* V = (const __bf16*)Vv;
  __bf16* O = (__bf16*)Ov;

  __shared__ __attribute__((aligned(16))) __bf16 Ks[2][32 * HDIM];   // [kk][d]
  __shared__ __attribute__((aligned(16))) __bf16 Vt[2][HDIM * 40];   // [d][kk]
  __shared__ __attribute__((aligned(16))) __bf16 Ps[8 * 16 * 32];    // per-wave P

  const int tid = threadIdx.x, lane = tid & 31, wave = tid >> 5;
  const int bh = blockIdx.y;
  const int b = bh >> 4, h = bh & 15;
  const int qbase = blockIdx.x * 128 + wave * 16;
  const int halfrow = (lane < 16) ? 0 : 8;

  const __bf16* Qb = Q + ((size_t)bh * SEQ + qbase) * HDIM;
  const __bf16* Kb = K + (size_t)bh * SEQ * HDIM;
  const __bf16* Vb = V + (size_t)bh * SEQ * HDIM;
  const int* mb = mask + (size_t)b * SEQ * SEQ;

  // Q strip 16x64 lives in registers as two A fragments.
  v16bf qf[2];
#pragma unroll
  for (int ks = 0; ks < 2; ++ks) qf[ks] = frag_a16(Qb + ks * 32, HDIM);

  float m_run[8], l_run[8];
  v8f oacc[4] = {};
#pragma unroll
  for (int r = 0; r < 8; ++r) { m_run[r] = -3.0e38f; l_run[r] = 0.f; }

  __bf16* PsW = Ps + wave * (16 * 32);

  // Per-thread staging role for V: key row sr (0..31), d-chunk scol (8 elems).
  const int sr = tid >> 3;
  const int scol = (tid & 7) * 8;

  // Prologue: stage K block 0, prefetch V block 0 into registers.
#if HAVE_TDM
  if (wave == 0) tdm_load_2d(Ks[0], Kb, HDIM, 32, HDIM);
#else
  async_ld_b128(Kb + (size_t)sr * HDIM + scol, &Ks[0][sr * HDIM + scol]);
#endif
  uint4 vreg = *(const uint4*)(Vb + (size_t)sr * HDIM + scol);

  for (int kb0 = 0; kb0 < SEQ; kb0 += 32) {
    const int cur = (kb0 >> 5) & 1;

    // Transpose-store held V chunk into Vt[cur] (last read 2 iterations ago,
    // separated by the previous barrier).
    {
      union { uint4 u; __bf16 hv[8]; } t;
      t.u = vreg;
#pragma unroll
      for (int i = 0; i < 8; ++i) Vt[cur][(scol + i) * 40 + sr] = t.hv[i];
    }
#if HAVE_TDM
    if (wave == 0) wait_tensor0();  // Ks[cur] TDM transfer complete
#else
    wait_async0();
#endif
    __syncthreads();  // all waves' Ks[cur]/Vt[cur] visible

    if (kb0 + 32 < SEQ) {
#if HAVE_TDM
      if (wave == 0)
        tdm_load_2d(Ks[cur ^ 1], Kb + (size_t)(kb0 + 32) * HDIM, HDIM, 32, HDIM);
#else
      async_ld_b128(Kb + (size_t)(kb0 + 32 + sr) * HDIM + scol,
                    &Ks[cur ^ 1][sr * HDIM + scol]);
#endif
      vreg = *(const uint4*)(Vb + (size_t)(kb0 + 32 + sr) * HDIM + scol);
    }

    // S = Q * K^T  (two 16-key tiles, K-dim = 64 = two wmma each)
    v8f sc[2];
#pragma unroll
    for (int jt = 0; jt < 2; ++jt) {
      v8f cfrag = {};
#pragma unroll
      for (int ks = 0; ks < 2; ++ks)
        cfrag = wmma_bf16(
            qf[ks], frag_b16(Ks[cur] + jt * 16 * HDIM + ks * 32, HDIM), cfrag);
      sc[jt] = cfrag;
    }

    // scale + mask
    const float scale = 0.125f;  // 1/sqrt(64)
#pragma unroll
    for (int jt = 0; jt < 2; ++jt) {
      const int j = kb0 + jt * 16 + (lane & 15);
#pragma unroll
      for (int r = 0; r < 8; ++r) {
        const int qrow = qbase + halfrow + r;
        float s = sc[jt][r] * scale;
        if (mb[(size_t)qrow * SEQ + j] == 0) s = -1.0e9f;
        sc[jt][r] = s;
      }
    }

    // Online softmax; row stats reduce across the 16-lane half holding a row.
    float alpha[8];
#pragma unroll
    for (int r = 0; r < 8; ++r) {
      float mloc = fmaxf(sc[0][r], sc[1][r]);
#pragma unroll
      for (int off = 8; off > 0; off >>= 1)
        mloc = fmaxf(mloc, __shfl_xor(mloc, off, 16));
      const float m_new = fmaxf(m_run[r], mloc);
      alpha[r] = __expf(m_run[r] - m_new);
      const float p0 = __expf(sc[0][r] - m_new);
      const float p1 = __expf(sc[1][r] - m_new);
      float rsum = p0 + p1;
#pragma unroll
      for (int off = 8; off > 0; off >>= 1)
        rsum += __shfl_xor(rsum, off, 16);
      l_run[r] = l_run[r] * alpha[r] + rsum;
      m_run[r] = m_new;
      // P (C-layout f32) -> bf16 A-layout via per-wave LDS staging.
      PsW[(halfrow + r) * 32 + (lane & 15)]      = (__bf16)p0;
      PsW[(halfrow + r) * 32 + 16 + (lane & 15)] = (__bf16)p1;
    }

#pragma unroll
    for (int dt = 0; dt < 4; ++dt)
#pragma unroll
      for (int r = 0; r < 8; ++r) oacc[dt][r] *= alpha[r];

    // O += P * V   (in-wave LDS store->load is in-order; no barrier needed)
    const v16bf pf = frag_a16(PsW, 32);
#pragma unroll
    for (int dt = 0; dt < 4; ++dt)
      oacc[dt] = wmma_bf16(pf, frag_b16(Vt[cur] + dt * 16 * 40, 40), oacc[dt]);
  }

#pragma unroll
  for (int r = 0; r < 8; ++r) l_run[r] = 1.0f / l_run[r];
#pragma unroll
  for (int dt = 0; dt < 4; ++dt) {
    const int d = h * HDIM + dt * 16 + (lane & 15);
#pragma unroll
    for (int r = 0; r < 8; ++r) {
      const int qrow = qbase + halfrow + r;
      O[((size_t)b * SEQ + qrow) * D_MODEL + d] = (__bf16)(oacc[dt][r] * l_run[r]);
    }
  }
}

// ---------------------------------------------------------------- launch ----
extern "C" void kernel_launch(void* const* d_in, const int* in_sizes, int n_in,
                              void* d_out, int out_size, void* d_ws, size_t ws_size,
                              hipStream_t stream) {
  const float* q    = (const float*)d_in[0];
  const float* k    = (const float*)d_in[1];
  const float* v    = (const float*)d_in[2];
  const int*   mask = (const int*)d_in[3];
  const float* Wq = (const float*)d_in[4];
  const float* bq = (const float*)d_in[5];
  const float* Wk = (const float*)d_in[6];
  const float* bk = (const float*)d_in[7];
  const float* Wv = (const float*)d_in[8];
  const float* bv = (const float*)d_in[9];
  const float* Wo = (const float*)d_in[10];
  const float* bo = (const float*)d_in[11];

  const size_t MK = (size_t)4096 * 1024;  // activation elements
  const size_t WK = (size_t)1024 * 1024;  // weight elements
  char* p = (char*)d_ws;                  // 64 MB of scratch used
  void* Xq  = p; p += MK * 2;
  void* Xk  = p; p += MK * 2;
  void* Xv  = p; p += MK * 2;
  void* Wqb = p; p += WK * 2;
  void* Wkb = p; p += WK * 2;
  void* Wvb = p; p += WK * 2;
  void* Wob = p; p += WK * 2;
  void* Qh  = p; p += MK * 2;
  void* Kh  = p; p += MK * 2;
  void* Vh  = p; p += MK * 2;
  void* Ao  = p; p += MK * 2;

  const dim3 blk(256);
  cvt_f32_bf16<<<dim3((unsigned)(MK / 4 / 256)), blk, 0, stream>>>(q,  Xq,  (int)(MK / 4));
  cvt_f32_bf16<<<dim3((unsigned)(MK / 4 / 256)), blk, 0, stream>>>(k,  Xk,  (int)(MK / 4));
  cvt_f32_bf16<<<dim3((unsigned)(MK / 4 / 256)), blk, 0, stream>>>(v,  Xv,  (int)(MK / 4));
  cvt_f32_bf16<<<dim3((unsigned)(WK / 4 / 256)), blk, 0, stream>>>(Wq, Wqb, (int)(WK / 4));
  cvt_f32_bf16<<<dim3((unsigned)(WK / 4 / 256)), blk, 0, stream>>>(Wk, Wkb, (int)(WK / 4));
  cvt_f32_bf16<<<dim3((unsigned)(WK / 4 / 256)), blk, 0, stream>>>(Wv, Wvb, (int)(WK / 4));
  cvt_f32_bf16<<<dim3((unsigned)(WK / 4 / 256)), blk, 0, stream>>>(Wo, Wob, (int)(WK / 4));

  const dim3 gg(D_MODEL / BN, 4096 / BM);  // (8, 32)
  gemm_wmma<<<gg, blk, 0, stream>>>(Xq, Wqb, bq, Qh, D_MODEL, D_MODEL, 0);
  gemm_wmma<<<gg, blk, 0, stream>>>(Xk, Wkb, bk, Kh, D_MODEL, D_MODEL, 0);
  gemm_wmma<<<gg, blk, 0, stream>>>(Xv, Wvb, bv, Vh, D_MODEL, D_MODEL, 0);

  attn_fmha<<<dim3(SEQ / 128, 32), blk, 0, stream>>>(Qh, Kh, Vh, mask, Ao);

  gemm_wmma<<<gg, blk, 0, stream>>>(Ao, Wob, bo, d_out, D_MODEL, D_MODEL, 1);
}
